// ColumnParallelQuantizedLinear_22058952032259
// MI455X (gfx1250) — matrix operand order repaired and checked by reading, not compiled, not run on hardware
//
#include <hip/hip_runtime.h>

typedef __attribute__((ext_vector_type(16))) _Float16 v16h;
typedef __attribute__((ext_vector_type(8)))  _Float16 v8h;
typedef __attribute__((ext_vector_type(2)))  _Float16 h2;
typedef __attribute__((ext_vector_type(8)))  float    v8f;

#define M_DIM   64
#define K_DIM   4096
#define N_DIM   11008
#define GROUP   128
#define K_TILE  128                // one quant group per K tile
#define BLOCK_N 64
#define KTILES  (K_DIM / K_TILE)   // 32
#define LDS_STRIDE (K_TILE + 16)   // f16 elems; 288-byte rows (32B aligned)

// out[64,11008] = x[64,4096] * dequant(qweight)[4096,11008] + bias
//
// K is re-permuted within each group of 8 (j <-> j+4 interleave) identically on
// A and B, so dequant can emit packed f16 pairs straight from nibble bit-tricks
// (v_pk_add_f16 / v_pk_mul_f16) and A staging emits the same order for free via
// v_cvt_pk_f16_f32(lo[j], hi[j]).
//
// Raw qweight tiles (16x64 u32 = 4KB) are double-buffered in LDS via
// global_load_async_to_lds_b128 (ASYNCcnt), issued one tile ahead to overlap
// the dominant HBM stream (22.5 MB) with dequant + WMMA.
__global__ __launch_bounds__(256)
void ColumnParallelQuantizedLinear_wmma(const float* __restrict__ x,
                                        const unsigned int* __restrict__ qweight,
                                        const unsigned int* __restrict__ qzeros,
                                        const float* __restrict__ scales,
                                        const float* __restrict__ bias,
                                        float* __restrict__ out)
{
    __shared__ _Float16     Alds[M_DIM][LDS_STRIDE];    // A tile (K-permuted), m-major
    __shared__ _Float16     Blds[BLOCK_N][LDS_STRIDE];  // B tile (K-permuted), n-major
    __shared__ unsigned int Braw[2][16][BLOCK_N];       // raw qweight tiles (double buffer)

    const int tid  = threadIdx.x;
    const int lane = tid & 31;      // wave32
    const int wave = tid >> 5;      // 0..7
    const int wm   = wave & 3;      // M group: rows wm*16 .. +15
    const int wn   = wave >> 2;     // N half:  cols wn*32 .. +31
    const int n0   = blockIdx.x * BLOCK_N;

    // Async raw-tile copy: 256 threads x 16B lanes cover 16 rows x 256B.
    auto issue_async = [&](int kt, int buf) {
        const int row = tid >> 4;        // 0..15 packed row
        const int c16 = tid & 15;        // 16-byte chunk within 256B row
        const unsigned long long gsrc = (unsigned long long)
            (qweight + (size_t)((kt * K_TILE) / 8 + row) * N_DIM + n0 + c16 * 4);
        const unsigned ldst = (unsigned)(size_t)&Braw[buf][row][c16 * 4];
        asm volatile("global_load_async_to_lds_b128 %0, %1, off"
                     :: "v"(ldst), "v"(gsrc) : "memory");
    };

    v8f acc0 = {};
    v8f acc1 = {};

    issue_async(0, 0);   // prologue: raw tile 0 in flight

    // Per-thread dequant column is fixed: nn = tid&63, rows rr = (tid>>6)+{0,4,8,12}
    const int nn    = tid & 63;
    const int rbase = tid >> 6;
    const int nq    = n0 + nn;

    for (int kt = 0; kt < KTILES; ++kt) {
        const int k0  = kt * K_TILE;
        const int buf = kt & 1;

        // ---- stage A tile: fp32 -> f16 (K-permuted pairs) ----
        for (int i = tid; i < M_DIM * (K_TILE / 8); i += 256) {
            const int row = i >> 4;          // K_TILE/8 == 16 chunks per row
            const int c8  = i & 15;
            const float4 lo = *(const float4*)(x + row * K_DIM + k0 + c8 * 8);
            const float4 hi = *(const float4*)(x + row * K_DIM + k0 + c8 * 8 + 4);
            v8h h;                           // pairs (k+j, k+j+4) -> v_cvt_pk_f16_f32
            h[0] = (_Float16)lo.x; h[1] = (_Float16)hi.x;
            h[2] = (_Float16)lo.y; h[3] = (_Float16)hi.y;
            h[4] = (_Float16)lo.z; h[5] = (_Float16)hi.z;
            h[6] = (_Float16)lo.w; h[7] = (_Float16)hi.w;
            *(v8h*)&Alds[row][c8 * 8] = h;
        }

        // ---- wait for this tile's raw copy; launch next tile's copy ----
#if __has_builtin(__builtin_amdgcn_s_wait_asynccnt)
        __builtin_amdgcn_s_wait_asynccnt(0);
#else
        asm volatile("s_wait_asynccnt 0x0" ::: "memory");
#endif
        if (kt + 1 < KTILES) issue_async(kt + 1, buf ^ 1);

        // ---- dequant raw[buf] -> Blds with packed f16 math ----
        {
            const int   g  = kt;                         // K_TILE == GROUP
            const float s  = scales[g * N_DIM + nq];
            const unsigned zq =
                (qzeros[g * (N_DIM / 8) + (nq >> 3)] >> ((nq & 7) * 4)) & 15u;
            const _Float16 sh = (_Float16)s;
            h2 s2; s2[0] = sh; s2[1] = sh;
            const unsigned hzd = (0x6400u + zq + 1u) * 0x00010001u; // f16 1024+zq+1, packed
            const h2 hz = __builtin_bit_cast(h2, hzd);
            #pragma unroll
            for (int t = 0; t < 4; ++t) {
                const int rr = rbase + t * 4;            // packed row (k = rr*8..+7)
                const unsigned w = Braw[buf][rr][nn];
                v8h hvec;
                #pragma unroll
                for (int j = 0; j < 4; ++j) {
                    // {1024+q_j, 1024+q_{j+4}} exact in f16
                    const unsigned tb =
                        ((w >> (4 * j)) & 0x000F000Fu) | 0x64006400u;
                    const h2 hv = __builtin_bit_cast(h2, tb);
                    const h2 r  = (hv - hz) * s2;        // v_pk_add_f16 + v_pk_mul_f16
                    hvec[2 * j]     = r[0];
                    hvec[2 * j + 1] = r[1];
                }
                *(v8h*)&Blds[nn][rr * 8] = hvec;         // ds_store_b128
            }
        }

        __syncthreads();

        // ---- 8 WMMAs over this K tile (4 k-chunks x 2 N sub-tiles) ----
        #pragma unroll
        for (int kk = 0; kk < K_TILE; kk += 32) {
            // A fragment: lane<16 holds K 0-15, lane>=16 holds K 16-31
            const _Float16* ap = &Alds[wm * 16 + (lane & 15)][kk + ((lane >> 4) * 16)];
            const v8h alo = *(const v8h*)ap;
            const v8h ahi = *(const v8h*)(ap + 8);
            v16h a;
            #pragma unroll
            for (int j = 0; j < 8; ++j) { a[j] = alo[j]; a[8 + j] = ahi[j]; }

            // B fragments: lane holds 16 contiguous K for column N = lane&15
            const _Float16* bp0 = &Blds[wn * 32 + (lane & 15)][kk + ((lane >> 4) * 16)];
            const v16h b0 = *(const v16h*)bp0;
            const v16h b1 = *(const v16h*)(bp0 + 16 * LDS_STRIDE);

            acc0 = __builtin_amdgcn_wmma_f32_16x16x32_f16(
                       false, a, false, b0, (short)0, acc0, false, false);
            acc1 = __builtin_amdgcn_wmma_f32_16x16x32_f16(
                       false, a, false, b1, (short)0, acc1, false, false);
        }

        __syncthreads();
    }

    // ---- epilogue: bias add + store (C layout: VGPR r -> M=r or r+8; col = lane&15) ----
    const int   col0 = n0 + wn * 32 + (lane & 15);
    const int   col1 = col0 + 16;
    const float bv0  = bias[col0];
    const float bv1  = bias[col1];
    const int   mrow = wm * 16 + ((lane >> 4) * 8);
    #pragma unroll
    for (int r = 0; r < 8; ++r) {
        out[(mrow + r) * N_DIM + col0] = acc0[r] + bv0;
        out[(mrow + r) * N_DIM + col1] = acc1[r] + bv1;
    }
}

extern "C" void kernel_launch(void* const* d_in, const int* in_sizes, int n_in,
                              void* d_out, int out_size, void* d_ws, size_t ws_size,
                              hipStream_t stream) {
    const float*        xx      = (const float*)d_in[0];
    const unsigned int* qweight = (const unsigned int*)d_in[1];
    const unsigned int* qzeros  = (const unsigned int*)d_in[2];
    const float*        scales  = (const float*)d_in[3];
    const float*        bias    = (const float*)d_in[4];
    float*              out     = (float*)d_out;

    dim3 grid(N_DIM / BLOCK_N);  // 172 blocks
    dim3 block(256);             // 8 wave32
    hipLaunchKernelGGL(ColumnParallelQuantizedLinear_wmma, grid, block, 0, stream,
                       xx, qweight, qzeros, scales, bias, out);
}
